// FastNeuron_58944131170734
// MI455X (gfx1250) — matrix operand order repaired and checked by reading, not compiled, not run on hardware
//
#include <hip/hip_runtime.h>
#include <math.h>

#define Bb 32
#define Ss 2048
#define Dd 2560
#define Rr 64
#define DQq 128
#define DVv 512
#define DPp 128
#define DCc 128
#define DRr 128
#define DHh 384

typedef __attribute__((ext_vector_type(16))) _Float16 v16h;
typedef __attribute__((ext_vector_type(8)))  float    v8f;

__device__ __forceinline__ float gelu_f(float x){ return 0.5f*x*(1.0f+erff(x*0.7071067811865475f)); }
__device__ __forceinline__ float sigm_f(float x){ return 1.0f/(1.0f+__expf(-x)); }
__device__ __forceinline__ float softplus_f(float x){ return (x>20.0f)?x:log1pf(__expf(x)); }

// ---------------------------------------------------------------------------
// WMMA GEMM: C(32xN) = [bias or C] + A(32xK) @ W(KxN)   (K%32==0, N%16==0)
// One wave per block computes a 16x16 tile via v_wmma_f32_16x16x32_f16.
// TRANSB is a compile-time flag so the inner loop is branch-free:
//   A operand: one contiguous 16-float run per lane  -> 4x float4 loads
//   B operand (transB): one contiguous 16-float run  -> 4x float4 loads
//   B operand (normal): 16 ldw-strided scalar loads
// ---------------------------------------------------------------------------
template<int TRANSB>
__global__ void k_gemm_wmma(const float* __restrict__ A, int lda,
                            const float* __restrict__ W, int ldw,
                            float* __restrict__ C, int ldc, int K,
                            const float* __restrict__ bias, int accumulate)
{
  const int lane = threadIdx.x;           // 0..31, wave32
  const int tn = blockIdx.x << 4;
  const int tm = blockIdx.y << 4;
  const int half = lane >> 4;
  const int l15  = lane & 15;
  v8f acc;
  if (accumulate) {
#pragma unroll
    for (int i=0;i<8;i++) acc[i] = C[(size_t)(tm + i + half*8)*ldc + tn + l15];
  } else {
    float bv = bias ? bias[tn + l15] : 0.0f;
#pragma unroll
    for (int i=0;i<8;i++) acc[i] = bv;
  }
  // A: lanes 0-15 hold K 0..15; lanes 16-31 hold K 16..31
  const float* Ap = A + (size_t)(tm + l15)*lda + half*16;
  // B: lanes 0-15 hold K 0..15 of column tn+l15; lanes 16-31 hold K 16..31
  const float* Wp = TRANSB ? (W + (size_t)(tn + l15)*ldw + half*16)
                           : (W + (size_t)(half*16)*ldw + tn + l15);
  for (int k0 = 0; k0 < K; k0 += 32) {
    float4 a0 = *(const float4*)(Ap +  0);
    float4 a1 = *(const float4*)(Ap +  4);
    float4 a2 = *(const float4*)(Ap +  8);
    float4 a3 = *(const float4*)(Ap + 12);
    v16h a;
    a[0]=(_Float16)a0.x; a[1]=(_Float16)a0.y; a[2]=(_Float16)a0.z; a[3]=(_Float16)a0.w;
    a[4]=(_Float16)a1.x; a[5]=(_Float16)a1.y; a[6]=(_Float16)a1.z; a[7]=(_Float16)a1.w;
    a[8]=(_Float16)a2.x; a[9]=(_Float16)a2.y; a[10]=(_Float16)a2.z; a[11]=(_Float16)a2.w;
    a[12]=(_Float16)a3.x; a[13]=(_Float16)a3.y; a[14]=(_Float16)a3.z; a[15]=(_Float16)a3.w;
    v16h bm;
    if (TRANSB) {
      float4 b0 = *(const float4*)(Wp +  0);
      float4 b1 = *(const float4*)(Wp +  4);
      float4 b2 = *(const float4*)(Wp +  8);
      float4 b3 = *(const float4*)(Wp + 12);
      bm[0]=(_Float16)b0.x; bm[1]=(_Float16)b0.y; bm[2]=(_Float16)b0.z; bm[3]=(_Float16)b0.w;
      bm[4]=(_Float16)b1.x; bm[5]=(_Float16)b1.y; bm[6]=(_Float16)b1.z; bm[7]=(_Float16)b1.w;
      bm[8]=(_Float16)b2.x; bm[9]=(_Float16)b2.y; bm[10]=(_Float16)b2.z; bm[11]=(_Float16)b2.w;
      bm[12]=(_Float16)b3.x; bm[13]=(_Float16)b3.y; bm[14]=(_Float16)b3.z; bm[15]=(_Float16)b3.w;
      Wp += 32;
    } else {
#pragma unroll
      for (int e=0;e<16;e++) bm[e] = (_Float16)Wp[(size_t)e*ldw];
      Wp += (size_t)32*ldw;
    }
    Ap += 32;
    acc = __builtin_amdgcn_wmma_f32_16x16x32_f16(false, a, false, bm,
                                                 (short)0, acc, false, false);
  }
#pragma unroll
  for (int i=0;i<8;i++) C[(size_t)(tm + i + half*8)*ldc + tn + l15] = acc[i];
}

// ---------------------------------------------------------------------------
// h_avg[b,d] = mean_s h[b,s,d]   (float4-coalesced column reduction)
// ---------------------------------------------------------------------------
__global__ void k_havg(const float* __restrict__ h, float* __restrict__ havg){
  int b = blockIdx.x;
  int c = blockIdx.y*128 + threadIdx.x;            // float4 column 0..639
  const float4* hp = (const float4*)(h + (size_t)b*Ss*Dd);
  float4 acc = make_float4(0.f,0.f,0.f,0.f);
  for (int s=0;s<Ss;s++){
    float4 v = hp[(size_t)s*(Dd/4) + c];
    acc.x+=v.x; acc.y+=v.y; acc.z+=v.z; acc.w+=v.w;
  }
  const float inv = 1.0f/(float)Ss;
  acc.x*=inv; acc.y*=inv; acc.z*=inv; acc.w*=inv;
  ((float4*)(havg + (size_t)b*Dd))[c] = acc;
}

// ---------------------------------------------------------------------------
// slots[b,r,:] = LN(mem_A[b,:,r]) * g + beta    grid(R, B), block 256
// ---------------------------------------------------------------------------
__global__ void k_slotln(const float* __restrict__ memA, const float* __restrict__ g,
                         const float* __restrict__ bta, float* __restrict__ slots){
  int r = blockIdx.x, b = blockIdx.y;
  const float* base = memA + (size_t)b*Dd*Rr + r;
  __shared__ float s1[256], s2[256];
  __shared__ float smean, srs;
  float sum=0.f, sq=0.f;
  for (int d=threadIdx.x; d<Dd; d+=256){ float x = base[(size_t)d*Rr]; sum+=x; sq+=x*x; }
  s1[threadIdx.x]=sum; s2[threadIdx.x]=sq; __syncthreads();
  for (int s=128;s>0;s>>=1){
    if (threadIdx.x<s){ s1[threadIdx.x]+=s1[threadIdx.x+s]; s2[threadIdx.x]+=s2[threadIdx.x+s]; }
    __syncthreads();
  }
  if (threadIdx.x==0){
    float m = s1[0]*(1.0f/Dd); float v = s2[0]*(1.0f/Dd) - m*m;
    smean=m; srs=rsqrtf(fmaxf(v,0.f)+1e-5f);
  }
  __syncthreads();
  float m=smean, rs=srs;
  float* out = slots + ((size_t)b*Rr + r)*Dd;
  for (int d=threadIdx.x; d<Dd; d+=256)
    out[d] = (base[(size_t)d*Rr]-m)*rs*g[d] + bta[d];
}

// ---------------------------------------------------------------------------
// scores + softmax + attention-weighted slot sum.  block per b, 256 threads.
// ---------------------------------------------------------------------------
__global__ void k_attn(const float* __restrict__ slots, const float* __restrict__ qk,
                       float* __restrict__ wslot){
  const int b = blockIdx.x;
  __shared__ float red[256];
  __shared__ float sc[Rr];
  const float* qkb = qk + (size_t)b*Dd;
  for (int r=0;r<Rr;r++){
    const float* sl = slots + ((size_t)b*Rr + r)*Dd;
    float p=0.f;
    for (int d=threadIdx.x; d<Dd; d+=256) p += sl[d]*qkb[d];
    red[threadIdx.x]=p; __syncthreads();
    for (int s=128;s>0;s>>=1){ if(threadIdx.x<s) red[threadIdx.x]+=red[threadIdx.x+s]; __syncthreads(); }
    if (threadIdx.x==0) sc[r] = red[0]*0.08838834764831845f;   // 1/sqrt(128)
    __syncthreads();
  }
  if (threadIdx.x==0){
    float mx=-1e30f;
    for (int r=0;r<Rr;r++) mx = fmaxf(mx, sc[r]);
    float s=0.f;
    for (int r=0;r<Rr;r++){ sc[r]=__expf(sc[r]-mx); s+=sc[r]; }
    float inv=1.0f/s;
    for (int r=0;r<Rr;r++) sc[r]*=inv;
  }
  __syncthreads();
  for (int d=threadIdx.x; d<Dd; d+=256){
    float acc=0.f;
    for (int r=0;r<Rr;r++) acc += sc[r]*slots[((size_t)b*Rr+r)*Dd + d];
    wslot[(size_t)b*Dd + d] = acc;
  }
}

__global__ void k_sub(const float* __restrict__ a, const float* __restrict__ b,
                      float* __restrict__ o, int n){
  int i = blockIdx.x*blockDim.x + threadIdx.x;
  if (i<n) o[i] = a[i]-b[i];
}

__global__ void k_gelu(float* __restrict__ x, int n){
  int i = blockIdx.x*blockDim.x + threadIdx.x;
  if (i<n) x[i] = gelu_f(x[i]);
}

// C[b,j] += s[b] * row[j]   (rank-1 fixup for scalar concat columns)
__global__ void k_axpy_row(float* __restrict__ C, int ldc, const float* __restrict__ s,
                           const float* __restrict__ row, int n){
  int i = blockIdx.x*blockDim.x + threadIdx.x;
  int total = Bb*n;
  if (i<total){ int b=i/n, j=i-b*n; C[(size_t)b*ldc + j] += s[b]*row[j]; }
}

// out[b] = act( X[b,:]·w + bias )    grid(B), block 256
// act: 0=none 1=sigmoid 2=min(softplus,0.1)
__global__ void k_matvec(const float* __restrict__ X, int ldx, const float* __restrict__ w,
                         const float* __restrict__ biasp, float* __restrict__ out,
                         int K, int act){
  int b = blockIdx.x;
  __shared__ float red[256];
  float p=0.f;
  for (int k=threadIdx.x;k<K;k+=256) p += X[(size_t)b*ldx+k]*w[k];
  red[threadIdx.x]=p; __syncthreads();
  for (int s=128;s>0;s>>=1){ if(threadIdx.x<s) red[threadIdx.x]+=red[threadIdx.x+s]; __syncthreads(); }
  if (threadIdx.x==0){
    float v = red[0] + biasp[0];
    if (act==1) v = sigm_f(v);
    else if (act==2) v = fminf(softplus_f(v), 0.1f);
    out[b]=v;
  }
}

// down[b,p] = gelu( mem_read[b,:]·(W_down_base+W_down_mod[b])[:,p] )  grid(B), block 128
__global__ void k_down(const float* __restrict__ mr, const float* __restrict__ Wdb,
                       const float* __restrict__ Wdm, float* __restrict__ down){
  int b = blockIdx.x, p = threadIdx.x;
  const float* m = mr + (size_t)b*Dd;
  const float* md = Wdm + (size_t)b*Dd*DPp;
  float acc=0.f;
  for (int d=0; d<Dd; d++) acc += m[d]*(Wdb[(size_t)d*DPp+p] + md[(size_t)d*DPp+p]);
  down[(size_t)b*DPp+p] = gelu_f(acc);
}

// output[b,d] = mem_read[b,d] + down[b,:]·(W_up_base+W_up_mod[b])[:,d]  grid(10,B), 256
__global__ void k_proj(const float* __restrict__ down, const float* __restrict__ Wub,
                       const float* __restrict__ Wum, const float* __restrict__ mr,
                       float* __restrict__ outv){
  int b = blockIdx.y;
  int d = blockIdx.x*256 + threadIdx.x;
  const float* dn = down + (size_t)b*DPp;
  const float* md = Wum + (size_t)b*DPp*Dd;
  float acc=0.f;
  for (int p=0;p<DPp;p++) acc += dn[p]*(Wub[(size_t)p*Dd+d] + md[(size_t)p*Dd+d]);
  outv[(size_t)b*Dd+d] = mr[(size_t)b*Dd+d] + acc;
}

// F.normalize over a segment X[b, off:off+len]   grid(B), block 256
__global__ void k_normseg(float* __restrict__ X, int ld, int off, int len){
  int b = blockIdx.x;
  float* x = X + (size_t)b*ld + off;
  __shared__ float red[256];
  __shared__ float sinv;
  float p=0.f;
  for (int i=threadIdx.x;i<len;i+=256){ float v=x[i]; p+=v*v; }
  red[threadIdx.x]=p; __syncthreads();
  for (int s=128;s>0;s>>=1){ if(threadIdx.x<s) red[threadIdx.x]+=red[threadIdx.x+s]; __syncthreads(); }
  if (threadIdx.x==0) sinv = 1.0f/fmaxf(sqrtf(red[0]), 1e-12f);
  __syncthreads();
  float inv=sinv;
  for (int i=threadIdx.x;i<len;i+=256) x[i]*=inv;
}

// mem_A_new = mem_A + lr[b]*wval[b,d]*wkey[b,r]  (copy if chunk==0)
__global__ void k_memA_upd(const float* __restrict__ memA, const float* __restrict__ lr,
                           const float* __restrict__ wval, const float* __restrict__ wkey,
                           const int* __restrict__ chunk, float* __restrict__ out){
  size_t i = (size_t)blockIdx.x*blockDim.x + threadIdx.x;
  const size_t total = (size_t)Bb*Dd*Rr;
  if (i>=total) return;
  int r = (int)(i & (Rr-1));
  size_t bd = i >> 6;
  int d = (int)(bd % Dd);
  int b = (int)(bd / Dd);
  float v = memA[i];
  if (*chunk != 0) v += lr[b]*wval[(size_t)b*Dd+d]*wkey[b*Rr+r];
  out[i]=v;
}

// W_down_mod_new[b,d,p] = 0.95*mod + scale_b * d_pat[b,d]*d_addr[b,p]
__global__ void k_wdown_upd(const float* __restrict__ mod, const float* __restrict__ sur,
                            const float* __restrict__ plr, const float* __restrict__ rawd,
                            const int* __restrict__ chunk, float* __restrict__ out){
  size_t i = (size_t)blockIdx.x*blockDim.x + threadIdx.x;
  const size_t total = (size_t)Bb*Dd*DPp;
  if (i>=total) return;
  int p = (int)(i & (DPp-1));
  size_t bd = i >> 7;
  int d = (int)(bd % Dd);
  int b = (int)(bd / Dd);
  float v = mod[i];
  if (*chunk != 0){
    float sc = sigm_f(10.0f*sur[b])*plr[b];
    v = 0.95f*v + sc * rawd[(size_t)b*2688 + d] * rawd[(size_t)b*2688 + 2560 + p];
  }
  out[i]=v;
}

// W_up_mod_new[b,p,d] = 0.95*mod + scale_b * u_pat[b,p]*u_addr[b,d]
__global__ void k_wup_upd(const float* __restrict__ mod, const float* __restrict__ sur,
                          const float* __restrict__ plr, const float* __restrict__ rawu,
                          const int* __restrict__ chunk, float* __restrict__ out){
  size_t i = (size_t)blockIdx.x*blockDim.x + threadIdx.x;
  const size_t total = (size_t)Bb*DPp*Dd;
  if (i>=total) return;
  int d = (int)(i % Dd);
  size_t bp = i / Dd;
  int p = (int)(bp % DPp);
  int b = (int)(bp / DPp);
  float v = mod[i];
  if (*chunk != 0){
    float sc = sigm_f(10.0f*sur[b])*plr[b];
    v = 0.95f*v + sc * rawu[(size_t)b*2688 + p] * rawu[(size_t)b*2688 + 128 + d];
  }
  out[i]=v;
}

// h_new[b,s,:] = LN( h[b,s,:] + g[b]*output[b,:] )   block per (b,s), 128 thr x 5 float4
__global__ void k_hnew(const float* __restrict__ h, const float* __restrict__ g,
                       const float* __restrict__ outv, const float* __restrict__ lng,
                       const float* __restrict__ lnb, float* __restrict__ out){
  int bs = blockIdx.x;
  int b = bs >> 11;                      // /2048
  const float4* hp = (const float4*)(h    + (size_t)bs*Dd);
  const float4* op = (const float4*)(outv + (size_t)b*Dd);
  float gv = g[b];
  float4 y[5];
  float sum=0.f, sq=0.f;
#pragma unroll
  for (int i=0;i<5;i++){
    int c = threadIdx.x + i*128;
    float4 hv = hp[c], ov = op[c], yv;
    yv.x=hv.x+gv*ov.x; yv.y=hv.y+gv*ov.y; yv.z=hv.z+gv*ov.z; yv.w=hv.w+gv*ov.w;
    y[i]=yv;
    sum += yv.x+yv.y+yv.z+yv.w;
    sq  += yv.x*yv.x+yv.y*yv.y+yv.z*yv.z+yv.w*yv.w;
  }
  __shared__ float s1[128], s2[128];
  __shared__ float sm, sr;
  s1[threadIdx.x]=sum; s2[threadIdx.x]=sq; __syncthreads();
  for (int s=64;s>0;s>>=1){
    if(threadIdx.x<s){ s1[threadIdx.x]+=s1[threadIdx.x+s]; s2[threadIdx.x]+=s2[threadIdx.x+s]; }
    __syncthreads();
  }
  if (threadIdx.x==0){
    float m=s1[0]*(1.0f/Dd); float v=s2[0]*(1.0f/Dd)-m*m;
    sm=m; sr=rsqrtf(fmaxf(v,0.f)+1e-5f);
  }
  __syncthreads();
  float m=sm, rs=sr;
  float4* outp = (float4*)(out + (size_t)bs*Dd);
  const float4* gp=(const float4*)lng; const float4* bp=(const float4*)lnb;
#pragma unroll
  for (int i=0;i<5;i++){
    int c = threadIdx.x + i*128;
    float4 gg=gp[c], bb=bp[c], yv=y[i], o;
    o.x=(yv.x-m)*rs*gg.x+bb.x; o.y=(yv.y-m)*rs*gg.y+bb.y;
    o.z=(yv.z-m)*rs*gg.z+bb.z; o.w=(yv.w-m)*rs*gg.w+bb.w;
    outp[c]=o;
  }
}

// ---------------------------------------------------------------------------
extern "C" void kernel_launch(void* const* d_in, const int* in_sizes, int n_in,
                              void* d_out, int out_size, void* d_ws, size_t ws_size,
                              hipStream_t stream) {
  const float* h    = (const float*)d_in[0];
  const float* memA = (const float*)d_in[1];
  const float* Wdm  = (const float*)d_in[2];
  const float* Wum  = (const float*)d_in[3];
  const float* ctx  = (const float*)d_in[4];
  const float* sur_W1=(const float*)d_in[5];  const float* sur_b1=(const float*)d_in[6];
  const float* sur_W2=(const float*)d_in[7];  const float* sur_b2=(const float*)d_in[8];
  const float* wk_W1 =(const float*)d_in[9];  const float* wk_b1 =(const float*)d_in[10];
  const float* wk_W2 =(const float*)d_in[11]; const float* wk_b2 =(const float*)d_in[12];
  const float* wv_W1 =(const float*)d_in[13]; const float* wv_b1 =(const float*)d_in[14];
  const float* wv_W2 =(const float*)d_in[15]; const float* wv_b2 =(const float*)d_in[16];
  const float* lr_W1 =(const float*)d_in[17]; const float* lr_b1 =(const float*)d_in[18];
  const float* lr_W2 =(const float*)d_in[19]; const float* lr_b2 =(const float*)d_in[20];
  const float* sln_g =(const float*)d_in[21]; const float* sln_b =(const float*)d_in[22];
  const float* rq_W  =(const float*)d_in[23]; const float* rq_b  =(const float*)d_in[24];
  const float* W_K   =(const float*)d_in[25]; const float* W_V   =(const float*)d_in[26];
  const float* vup_W =(const float*)d_in[27]; const float* vup_b =(const float*)d_in[28];
  const float* Wdb   =(const float*)d_in[29]; const float* Wub   =(const float*)d_in[30];
  const float* pwd_W1=(const float*)d_in[31]; const float* pwd_b1=(const float*)d_in[32];
  const float* pwd_W2=(const float*)d_in[33]; const float* pwd_b2=(const float*)d_in[34];
  const float* pwu_W1=(const float*)d_in[35]; const float* pwu_b1=(const float*)d_in[36];
  const float* pwu_W2=(const float*)d_in[37]; const float* pwu_b2=(const float*)d_in[38];
  const float* plr_W1=(const float*)d_in[39]; const float* plr_b1=(const float*)d_in[40];
  const float* plr_W2=(const float*)d_in[41]; const float* plr_b2=(const float*)d_in[42];
  const float* g_W1  =(const float*)d_in[43]; const float* g_b1  =(const float*)d_in[44];
  const float* g_W2  =(const float*)d_in[45]; const float* g_b2  =(const float*)d_in[46];
  const float* ln_g  =(const float*)d_in[47]; const float* ln_b  =(const float*)d_in[48];
  const float* rep_W1=(const float*)d_in[49]; const float* rep_b1=(const float*)d_in[50];
  const float* rep_W2=(const float*)d_in[51]; const float* rep_b2=(const float*)d_in[52];
  const int*   chunk =(const int*)  d_in[53];

  float* out_h    = (float*)d_out;                          // B*S*D
  float* out_memA = out_h    + (size_t)Bb*Ss*Dd;            // B*D*R
  float* out_wdm  = out_memA + (size_t)Bb*Dd*Rr;            // B*D*DP
  float* out_wum  = out_wdm  + (size_t)Bb*Dd*DPp;           // B*DP*D
  float* out_rep  = out_wum  + (size_t)Bb*DPp*Dd;           // B*DR

  float* w = (float*)d_ws;
  auto alloc = [&](size_t n){ float* p = w; w += n; return p; };
  float* havg   = alloc((size_t)Bb*Dd);
  float* slots  = alloc((size_t)Bb*Rr*Dd);
  float* query  = alloc((size_t)Bb*DQq);
  float* qk     = alloc((size_t)Bb*Dd);
  float* wslot  = alloc((size_t)Bb*Dd);
  float* vbuf   = alloc((size_t)Bb*DVv);
  float* mread  = alloc((size_t)Bb*Dd);
  float* err    = alloc((size_t)Bb*Dd);
  float* hidA   = alloc((size_t)Bb*DHh);
  float* hidB   = alloc((size_t)Bb*DHh);
  float* hid_lr = alloc((size_t)Bb*192);
  float* hid_pl = alloc((size_t)Bb*64);
  float* sur    = alloc(Bb);
  float* wkey   = alloc((size_t)Bb*Rr);
  float* wval   = alloc((size_t)Bb*Dd);
  float* lrv    = alloc(Bb);
  float* rawd   = alloc((size_t)Bb*2688);
  float* rawu   = alloc((size_t)Bb*2688);
  float* plrv   = alloc(Bb);
  float* down   = alloc((size_t)Bb*DPp);
  float* outputv= alloc((size_t)Bb*Dd);
  float* gv     = alloc(Bb);
  (void)in_sizes; (void)n_in; (void)out_size; (void)ws_size;

  auto gemm = [&](const float* A, int lda, const float* Wp, int ldw, float* C, int ldc,
                  int N, int K, const float* bias, int acc){
    k_gemm_wmma<0><<<dim3(N/16, 2), 32, 0, stream>>>(A, lda, Wp, ldw, C, ldc, K, bias, acc);
  };
  auto gemmT = [&](const float* A, int lda, const float* Wp, int ldw, float* C, int ldc,
                   int N, int K, const float* bias, int acc){
    k_gemm_wmma<1><<<dim3(N/16, 2), 32, 0, stream>>>(A, lda, Wp, ldw, C, ldc, K, bias, acc);
  };
  auto gelu = [&](float* x, int n){ k_gelu<<<(n+255)/256, 256, 0, stream>>>(x, n); };
  auto axpy = [&](float* C, int ldc, const float* s, const float* row, int n){
    k_axpy_row<<<(Bb*n+255)/256, 256, 0, stream>>>(C, ldc, s, row, n);
  };

  // 1) h_avg
  k_havg<<<dim3(Bb,5), 128, 0, stream>>>(h, havg);
  // 2) slots = LN(mem_A^T)
  k_slotln<<<dim3(Rr,Bb), 256, 0, stream>>>(memA, sln_g, sln_b, slots);
  // 3) query = [h_avg|ctx] @ rq_W + rq_b
  gemm(havg, Dd, rq_W, DQq, query, DQq, DQq, Dd, rq_b, 0);
  gemm(ctx, DCc, rq_W + (size_t)Dd*DQq, DQq, query, DQq, DQq, DCc, nullptr, 1);
  // 4) qk = query @ W_K^T  (so scores = slots · qk)
  gemmT(query, DQq, W_K, DQq, qk, Dd, Dd, DQq, nullptr, 0);
  // 5) softmax attention + weighted slot sum
  k_attn<<<Bb, 256, 0, stream>>>(slots, qk, wslot);
  // 6) mem_read = (wslot @ W_V) @ vup_W + vup_b
  gemm(wslot, Dd, W_V, DVv, vbuf, DVv, DVv, Dd, nullptr, 0);
  gemm(vbuf, DVv, vup_W, Dd, mread, Dd, Dd, DVv, vup_b, 0);
  // 7) error, surprise
  k_sub<<<(Bb*Dd+255)/256, 256, 0, stream>>>(havg, mread, err, Bb*Dd);
  gemm(err, Dd, sur_W1, DHh, hidA, DHh, DHh, Dd, sur_b1, 0);
  gemm(ctx, DCc, sur_W1 + (size_t)Dd*DHh, DHh, hidA, DHh, DHh, DCc, nullptr, 1);
  gelu(hidA, Bb*DHh);
  k_matvec<<<Bb, 256, 0, stream>>>(hidA, DHh, sur_W2, sur_b2, sur, DHh, 1);
  // 8) wkey / wval MLPs  (input [h_avg, surprise, ctx], K split 2560 + 1 + 128)
  gemm(havg, Dd, wk_W1, DHh, hidA, DHh, DHh, Dd, wk_b1, 0);
  gemm(ctx, DCc, wk_W1 + (size_t)2561*DHh, DHh, hidA, DHh, DHh, DCc, nullptr, 1);
  axpy(hidA, DHh, sur, wk_W1 + (size_t)2560*DHh, DHh);
  gelu(hidA, Bb*DHh);
  gemm(hidA, DHh, wk_W2, Rr, wkey, Rr, Rr, DHh, wk_b2, 0);
  gemm(havg, Dd, wv_W1, DHh, hidB, DHh, DHh, Dd, wv_b1, 0);
  gemm(ctx, DCc, wv_W1 + (size_t)2561*DHh, DHh, hidB, DHh, DHh, DCc, nullptr, 1);
  axpy(hidB, DHh, sur, wv_W1 + (size_t)2560*DHh, DHh);
  gelu(hidB, Bb*DHh);
  gemm(hidB, DHh, wv_W2, Dd, wval, Dd, Dd, DHh, wv_b2, 0);
  // 9) lr MLP  (input [surprise, ctx], K split 1 + 128)
  gemm(ctx, DCc, lr_W1 + 192, 192, hid_lr, 192, 192, DCc, lr_b1, 0);
  axpy(hid_lr, 192, sur, lr_W1, 192);
  gelu(hid_lr, Bb*192);
  k_matvec<<<Bb, 256, 0, stream>>>(hid_lr, 192, lr_W2, lr_b2, lrv, 192, 2);
  // 10) mem_A_new
  k_memA_upd<<<(int)(((size_t)Bb*Dd*Rr + 255)/256), 256, 0, stream>>>(memA, lrv, wval, wkey, chunk, out_memA);
  // 11) down / projected / output
  k_down<<<Bb, DPp, 0, stream>>>(mread, Wdb, Wdm, down);
  k_proj<<<dim3(Dd/256, Bb), 256, 0, stream>>>(down, Wub, Wum, mread, outputv);
  // 12) pwd MLP  (input [h_avg, surprise, mem_read, ctx], K = 2560+1+2560+128)
  gemm(havg, Dd, pwd_W1, DHh, hidA, DHh, DHh, Dd, pwd_b1, 0);
  gemm(mread, Dd, pwd_W1 + (size_t)2561*DHh, DHh, hidA, DHh, DHh, Dd, nullptr, 1);
  gemm(ctx, DCc, pwd_W1 + (size_t)5121*DHh, DHh, hidA, DHh, DHh, DCc, nullptr, 1);
  axpy(hidA, DHh, sur, pwd_W1 + (size_t)2560*DHh, DHh);
  gelu(hidA, Bb*DHh);
  gemm(hidA, DHh, pwd_W2, 2688, rawd, 2688, 2688, DHh, pwd_b2, 0);
  k_normseg<<<Bb, 256, 0, stream>>>(rawd, 2688, 0, 2560);     // d_pat
  k_normseg<<<Bb, 256, 0, stream>>>(rawd, 2688, 2560, 128);   // d_addr
  // 13) pwu MLP
  gemm(havg, Dd, pwu_W1, DHh, hidB, DHh, DHh, Dd, pwu_b1, 0);
  gemm(mread, Dd, pwu_W1 + (size_t)2561*DHh, DHh, hidB, DHh, DHh, Dd, nullptr, 1);
  gemm(ctx, DCc, pwu_W1 + (size_t)5121*DHh, DHh, hidB, DHh, DHh, DCc, nullptr, 1);
  axpy(hidB, DHh, sur, pwu_W1 + (size_t)2560*DHh, DHh);
  gelu(hidB, Bb*DHh);
  gemm(hidB, DHh, pwu_W2, 2688, rawu, 2688, 2688, DHh, pwu_b2, 0);
  k_normseg<<<Bb, 256, 0, stream>>>(rawu, 2688, 0, 128);      // u_pat
  k_normseg<<<Bb, 256, 0, stream>>>(rawu, 2688, 128, 2560);   // u_addr
  // 14) plr MLP  (input [surprise, ctx], K = 1 + 128, hidden 64)
  gemm(ctx, DCc, plr_W1 + 64, 64, hid_pl, 64, 64, DCc, plr_b1, 0);
  axpy(hid_pl, 64, sur, plr_W1, 64);
  gelu(hid_pl, Bb*64);
  k_matvec<<<Bb, 256, 0, stream>>>(hid_pl, 64, plr_W2, plr_b2, plrv, 64, 2);
  // 15) W_mod updates
  k_wdown_upd<<<(int)(((size_t)Bb*Dd*DPp + 255)/256), 256, 0, stream>>>(Wdm, sur, plrv, rawd, chunk, out_wdm);
  k_wup_upd  <<<(int)(((size_t)Bb*DPp*Dd + 255)/256), 256, 0, stream>>>(Wum, sur, plrv, rawu, chunk, out_wum);
  // 16) gate g  (input [h_avg, output, ctx], K = 2560+2560+128)
  gemm(havg, Dd, g_W1, DHh, hidA, DHh, DHh, Dd, g_b1, 0);
  gemm(outputv, Dd, g_W1 + (size_t)2560*DHh, DHh, hidA, DHh, DHh, Dd, nullptr, 1);
  gemm(ctx, DCc, g_W1 + (size_t)5120*DHh, DHh, hidA, DHh, DHh, DCc, nullptr, 1);
  gelu(hidA, Bb*DHh);
  k_matvec<<<Bb, 256, 0, stream>>>(hidA, DHh, g_W2, g_b2, gv, DHh, 1);
  // 17) h_new = LN(h + g*output)
  k_hnew<<<Bb*Ss, 128, 0, stream>>>(h, gv, outputv, ln_g, ln_b, out_h);
  // 18) report MLP  (input [error, surprise, down, g], K = 2560+1+128+1)
  gemm(err, Dd, rep_W1, DHh, hidB, DHh, DHh, Dd, rep_b1, 0);
  gemm(down, DPp, rep_W1 + (size_t)2561*DHh, DHh, hidB, DHh, DHh, DPp, nullptr, 1);
  axpy(hidB, DHh, sur, rep_W1 + (size_t)2560*DHh, DHh);
  axpy(hidB, DHh, gv,  rep_W1 + (size_t)2689*DHh, DHh);
  gelu(hidB, Bb*DHh);
  gemm(hidB, DHh, rep_W2, DRr, out_rep, DRr, DRr, DHh, rep_b2, 0);
}
